// SelfAttentionLSTM8_56401510531409
// MI455X (gfx1250) — matrix operand order repaired and checked by reading, not compiled, not run on hardware
//
#include <hip/hip_runtime.h>
#include <hip/hip_bf16.h>

// SA-ConvLSTM for MI455X (gfx1250): bf16 WMMA GEMMs + fp32 state.
// B=16, T=16, C=128, H=W=32.  N = B*HW = 16384, S = B*C*HW = 2097152.

typedef __attribute__((ext_vector_type(16))) __bf16 v16bf;
typedef __attribute__((ext_vector_type(8)))  __bf16 v8bf;
typedef __attribute__((ext_vector_type(8)))  float  v8f;

#define S_ELEMS 2097152ull   // 16*128*1024

// ---------------------------------------------------------------------------
// GEMM:  D[m,n] = act( A[m,:] . Bact[:,n] + bias[m] )
//   A    : bf16 [M,K] row-major (weights), M = gridDim.y*128
//   Bact : bf16 channel-contiguous activations [b][p][kstride], element
//          (k, n) at Bact[(b*1024+p)*kstride + k],  n = b*1024+p
//   outF : planar fp32  out[matid*mstride + (b*128+c)*1024 + p]   (matid=m>>7)
//   outB : bf16 channel-contig out[(b*1024+p)*ostride + choff + c]
//   acts : 4-bit act code per matid (0 none, 1 sigmoid, 2 tanh)
// ---------------------------------------------------------------------------
__global__ __launch_bounds__(256)
void k_gemm(const __bf16* __restrict__ A, const float* __restrict__ bias,
            const __bf16* __restrict__ Bact, int K, int kstride,
            float* __restrict__ outF, unsigned long long mstride,
            __bf16* __restrict__ outB, int ostride, int choff,
            int acts)
{
    __shared__ __bf16 sA[128 * 40];   // padded: 40 halves/row -> bank-conflict-free
    __shared__ __bf16 sB[128 * 40];

    const int tid  = threadIdx.x;
    const int lane = tid & 31;
    const int w    = tid >> 5;
    const int wm   = w & 3;           // 4 waves along M
    const int wn   = w >> 2;          // 2 waves along N
    const int m0   = blockIdx.y * 128;
    const int n0   = blockIdx.x * 128;
    const int bidx = n0 >> 10;        // batch
    const int p0   = n0 & 1023;       // pixel base (tile stays inside one batch)

    const __bf16* Ag = A + (size_t)m0 * K;
    const __bf16* Bg = Bact + ((size_t)bidx * 1024 + p0) * (size_t)kstride;

    v8f acc[2][4] = {};

    const int srow  = tid >> 1;       // 0..127 staging row
    const int spart = (tid & 1) * 16; // half-chunk of 32 halves

    const int rsel = lane & 15;
    const int hi   = lane >> 4;

    for (int k0 = 0; k0 < K; k0 += 32) {
        __syncthreads();
        // ---- stage A (128 x 32 bf16) ----
        {
            const __bf16* src = Ag + (size_t)srow * K + k0 + spart;
            v8bf x0 = *(const v8bf*)(src);
            v8bf x1 = *(const v8bf*)(src + 8);
            *(v8bf*)(sA + srow * 40 + spart)     = x0;
            *(v8bf*)(sA + srow * 40 + spart + 8) = x1;
            if (k0 + 32 < K) __builtin_prefetch(src + 32, 0, 1);
        }
        // ---- stage B (128 n-rows x 32 k) ----
        {
            const __bf16* src = Bg + (size_t)srow * kstride + k0 + spart;
            v8bf x0 = *(const v8bf*)(src);
            v8bf x1 = *(const v8bf*)(src + 8);
            *(v8bf*)(sB + srow * 40 + spart)     = x0;
            *(v8bf*)(sB + srow * 40 + spart + 8) = x1;
            if (k0 + 32 < K) __builtin_prefetch(src + 32, 0, 1);
        }
        __syncthreads();

        // ---- fragments (ISA 7.12.2 16-bit layouts) ----
        v16bf af[2], bfr[4];
#pragma unroll
        for (int i = 0; i < 2; ++i) {
            // A 16x32: lane<16 -> K 0..7 & 16..23 ; lane>=16 -> K 8..15 & 24..31
            const __bf16* pa = sA + (wm * 32 + i * 16 + rsel) * 40;
            const int kb = hi * 8;
            v8bf lo = *(const v8bf*)(pa + kb);
            v8bf hh = *(const v8bf*)(pa + 16 + kb);
            af[i] = __builtin_shufflevector(lo, hh, 0,1,2,3,4,5,6,7,8,9,10,11,12,13,14,15);
        }
#pragma unroll
        for (int j = 0; j < 4; ++j) {
            // B 32x16: lane<16 -> col n, K 0..7 & 16..23 ; lane>=16 -> col n, K 8..15 & 24..31
            const __bf16* pb = sB + (wn * 64 + j * 16 + rsel) * 40;
            const int ko = hi * 8;
            v8bf lo = *(const v8bf*)(pb + ko);
            v8bf hh = *(const v8bf*)(pb + 16 + ko);
            bfr[j] = __builtin_shufflevector(lo, hh, 0,1,2,3,4,5,6,7,8,9,10,11,12,13,14,15);
        }
#pragma unroll
        for (int i = 0; i < 2; ++i)
#pragma unroll
            for (int j = 0; j < 4; ++j)
                acc[i][j] = __builtin_amdgcn_wmma_f32_16x16x32_bf16(
                    false, af[i], false, bfr[j], (short)0, acc[i][j], false, false);
    }

    // ---- epilogue: C/D layout: vgpr r, lane<16 -> M=r,N=lane ; lane>=16 -> M=8+r ----
    const int nl = lane & 15;
    const int mb = (lane >> 4) * 8;
#pragma unroll
    for (int i = 0; i < 2; ++i) {
#pragma unroll
        for (int j = 0; j < 4; ++j) {
            const int mt = wm * 32 + i * 16 + mb;
            const int pp = p0 + wn * 64 + j * 16 + nl;
#pragma unroll
            for (int r = 0; r < 8; ++r) {
                const int mrow  = m0 + mt + r;
                const int matid = mrow >> 7;
                const int c     = mrow & 127;
                float v = acc[i][j][r] + bias[mrow];
                const int ac = (acts >> (matid * 4)) & 7;
                if (ac == 1)      v = 1.0f / (1.0f + __expf(-v));
                else if (ac == 2) v = tanhf(v);
                if (outF) {
                    outF[(size_t)matid * mstride + ((size_t)bidx * 128 + c) * 1024 + pp] = v;
                } else {
                    outB[((size_t)bidx * 1024 + pp) * ostride + choff + c] = (__bf16)v;
                }
            }
        }
    }
}

// ---------------------------------------------------------------------------
// weights fp32 -> bf16
__global__ void k_cvt(const float* __restrict__ W5, const float* __restrict__ W8,
                      __bf16* __restrict__ w5b, __bf16* __restrict__ w8b)
{
    const int i = blockIdx.x * 256 + threadIdx.x;
    if (i < 5 * 128 * 128) w5b[i] = (__bf16)W5[i];
    if (i < 8 * 128 * 256) w8b[i] = (__bf16)W8[i];
}

// init: hx[H half]=0, hx[x half]=bf16(x_t0) (transposed to [b][p][c]),
//       m=0, mbf=0, cst=c0.   grid (16,4,32) = (b, c/32, p/32)
__global__ void k_init(const float* __restrict__ x, const float* __restrict__ c0,
                       __bf16* __restrict__ hx, float* __restrict__ m,
                       __bf16* __restrict__ mbf, float* __restrict__ cst)
{
    __shared__ float t[32][33];
    const int b = blockIdx.x, c0g = blockIdx.y * 32, p0g = blockIdx.z * 32;
    const int tid = threadIdx.x, pl = tid & 31, cq = tid >> 5;
#pragma unroll
    for (int it = 0; it < 4; ++it) {
        const int cl = cq + it * 8;
        const size_t sidx = ((size_t)b * 128 + c0g + cl) * 1024 + p0g + pl;
        t[cl][pl] = x[((size_t)b * 16 * 128 + c0g + cl) * 1024 + p0g + pl]; // t=0
        m[sidx]   = 0.0f;
        cst[sidx] = c0[sidx];
    }
    __syncthreads();
    const int clw = tid & 31, pq = tid >> 5;
#pragma unroll
    for (int it = 0; it < 4; ++it) {
        const int plw = pq + it * 8;
        const size_t base = (size_t)b * 1024 + p0g + plw;
        hx[base * 256 + c0g + clw]        = (__bf16)0.0f;
        hx[base * 256 + 128 + c0g + clw]  = (__bf16)t[clw][plw];
        mbf[base * 128 + c0g + clw]       = (__bf16)0.0f;
    }
}

// LSTM core: Cst = Cst*a + ga*gv ; Hmid = a1*tanh(Cst) -> hz[ch 0..127] bf16
__global__ void k_lstm(const float* __restrict__ g4, float* __restrict__ cst,
                       __bf16* __restrict__ hz)
{
    __shared__ __bf16 t[32][33];
    const int b = blockIdx.x, c0g = blockIdx.y * 32, p0g = blockIdx.z * 32;
    const int tid = threadIdx.x, pl = tid & 31, cq = tid >> 5;
#pragma unroll
    for (int it = 0; it < 4; ++it) {
        const int cl = cq + it * 8;
        const size_t idx = ((size_t)b * 128 + c0g + cl) * 1024 + p0g + pl;
        const float a  = g4[idx];
        const float ga = g4[S_ELEMS + idx];
        const float gv = g4[2 * S_ELEMS + idx];
        const float a1 = g4[3 * S_ELEMS + idx];
        const float cv = cst[idx] * a + ga * gv;
        cst[idx] = cv;
        t[cl][pl] = (__bf16)(a1 * tanhf(cv));
    }
    __syncthreads();
    const int clw = tid & 31, pq = tid >> 5;
#pragma unroll
    for (int it = 0; it < 4; ++it) {
        const int plw = pq + it * 8;
        hz[((size_t)b * 1024 + p0g + plw) * 256 + c0g + clw] = t[clw][plw];
    }
}

// spatial 32x32 transpose per (b,c): qt[..,i,j] = q[..,j,i].  grid 2048
__global__ void k_tq(const float* __restrict__ q, float* __restrict__ qt)
{
    __shared__ float t[32][33];
    const float* src = q  + (size_t)blockIdx.x * 1024;
    float*       dst = qt + (size_t)blockIdx.x * 1024;
    const int tid = threadIdx.x, j = tid & 31, iq = tid >> 5;
#pragma unroll
    for (int it = 0; it < 4; ++it) { const int i = iq + it * 8; t[i][j] = src[i * 32 + j]; }
    __syncthreads();
#pragma unroll
    for (int it = 0; it < 4; ++it) { const int i = iq + it * 8; dst[i * 32 + j] = t[j][i]; }
}

// softmax over last axis (32 wide, one wave32 per row) + z = concat(zh,zm) bf16
// att: [0]=vh [1]=kh [2]=qh_raw(unused) [3]=km [4]=vm.  grid (16,32,4)=(b,i,c/32)
__global__ void k_soft(const float* __restrict__ att, const float* __restrict__ qt,
                       __bf16* __restrict__ z)
{
    __shared__ __bf16 t[32][66];
    const int b = blockIdx.x, i = blockIdx.y, c0g = blockIdx.z * 32;
    const int tid = threadIdx.x, j = tid & 31, w = tid >> 5;
#pragma unroll
    for (int it = 0; it < 4; ++it) {
        const int cl = w * 4 + it;
        const size_t idx = ((size_t)b * 128 + c0g + cl) * 1024 + i * 32 + j;
        const float vh = att[idx];
        const float kh = att[S_ELEMS + idx];
        const float km = att[3 * S_ELEMS + idx];
        const float vm = att[4 * S_ELEMS + idx];
        const float qv = qt[idx];
        float s1 = kh * qv, s2 = qv * km;
        float m1 = s1, m2 = s2;
        for (int o = 16; o > 0; o >>= 1) {
            m1 = fmaxf(m1, __shfl_xor(m1, o, 32));
            m2 = fmaxf(m2, __shfl_xor(m2, o, 32));
        }
        float e1 = __expf(s1 - m1), e2 = __expf(s2 - m2);
        float d1 = e1, d2 = e2;
        for (int o = 16; o > 0; o >>= 1) {
            d1 += __shfl_xor(d1, o, 32);
            d2 += __shfl_xor(d2, o, 32);
        }
        t[j][cl]      = (__bf16)(vh * (e1 / d1));
        t[j][33 + cl] = (__bf16)(vm * (e2 / d2));
    }
    __syncthreads();
    for (int q2 = tid; q2 < 2048; q2 += 256) {
        const int jr = q2 >> 6, ch = q2 & 63;
        const __bf16 v = (ch < 32) ? t[jr][ch] : t[jr][33 + (ch - 32)];
        const int cg = (ch < 32) ? (c0g + ch) : (128 + c0g + ch - 32);
        z[((size_t)b * 1024 + i * 32 + jr) * 256 + cg] = v;
    }
}

// final: mt = gt*it + (1-it)*m ; ht = ot*mt ; pack ht,x_{t+1} -> hx, mt -> mbf
__global__ void k_final(const float* __restrict__ og, float* __restrict__ m,
                        const float* __restrict__ x, int tnext,
                        __bf16* __restrict__ hx, __bf16* __restrict__ mbf,
                        float* __restrict__ out)
{
    __shared__ __bf16 th[32][33], tm[32][33], tx[32][33];
    const int b = blockIdx.x, c0g = blockIdx.y * 32, p0g = blockIdx.z * 32;
    const int tid = threadIdx.x, pl = tid & 31, cq = tid >> 5;
#pragma unroll
    for (int it = 0; it < 4; ++it) {
        const int cl = cq + it * 8;
        const size_t idx = ((size_t)b * 128 + c0g + cl) * 1024 + p0g + pl;
        const float ot  = og[idx];
        const float gt  = og[S_ELEMS + idx];
        const float itg = og[2 * S_ELEMS + idx];
        const float mt  = gt * itg + (1.0f - itg) * m[idx];
        m[idx] = mt;
        const float ht = ot * mt;
        th[cl][pl] = (__bf16)ht;
        tm[cl][pl] = (__bf16)mt;
        if (tnext < 16)
            tx[cl][pl] = (__bf16)x[(((size_t)b * 16 + tnext) * 128 + c0g + cl) * 1024 + p0g + pl];
        if (out) out[idx] = ht;   // planar [B,C,H,W] == d_out layout
    }
    __syncthreads();
    const int clw = tid & 31, pq = tid >> 5;
#pragma unroll
    for (int it = 0; it < 4; ++it) {
        const int plw = pq + it * 8;
        const size_t base = (size_t)b * 1024 + p0g + plw;
        hx[base * 256 + c0g + clw]  = th[clw][plw];
        mbf[base * 128 + c0g + clw] = tm[clw][plw];
        if (tnext < 16) hx[base * 256 + 128 + c0g + clw] = tx[clw][plw];
    }
}

// ---------------------------------------------------------------------------
extern "C" void kernel_launch(void* const* d_in, const int* in_sizes, int n_in,
                              void* d_out, int out_size, void* d_ws, size_t ws_size,
                              hipStream_t stream)
{
    const float* x  = (const float*)d_in[0];
    const float* c0 = (const float*)d_in[1];
    const float* W5 = (const float*)d_in[2];
    const float* b5 = (const float*)d_in[3];
    const float* W8 = (const float*)d_in[4];
    const float* b8 = (const float*)d_in[5];
    float* out = (float*)d_out;

    const size_t S = S_ELEMS;
    char* p = (char*)d_ws;
    auto take = [&](size_t bytes) { char* r = p; p += (bytes + 255) & ~(size_t)255; return r; };

    __bf16* w5b = (__bf16*)take(5ull * 128 * 128 * 2);
    __bf16* w8b = (__bf16*)take(8ull * 128 * 256 * 2);
    __bf16* hx  = (__bf16*)take(16ull * 1024 * 256 * 2);  // [b][p][H | x]
    __bf16* hz  = (__bf16*)take(16ull * 1024 * 256 * 2);  // [b][p][Hmid | z]
    __bf16* zbf = (__bf16*)take(16ull * 1024 * 256 * 2);  // [b][p][zh | zm]
    __bf16* mbf = (__bf16*)take(16ull * 1024 * 128 * 2);
    float*  mfp = (float*) take(S * 4);
    float*  cst = (float*) take(S * 4);
    float*  g4  = (float*) take(4 * S * 4);   // gates; reused as ot/gt/it
    float*  att = (float*) take(5 * S * 4);   // vh kh qh km vm
    float*  qt  = (float*) take(S * 4);

    auto ACT = [](int a0, int a1, int a2, int a3) {
        return a0 | (a1 << 4) | (a2 << 8) | (a3 << 12);
    };

    const dim3 blk(256);
    const dim3 gEW(16, 4, 32);

    k_cvt<<<1024, blk, 0, stream>>>(W5, W8, w5b, w8b);
    k_init<<<gEW, blk, 0, stream>>>(x, c0, hx, mfp, mbf, cst);

    for (int t = 0; t < 16; ++t) {
        // conv10..13: [512,256] x hx -> a_xh(sig), ga(sig), gv(tanh), a_xh1(sig)
        k_gemm<<<dim3(128, 4), blk, 0, stream>>>(w8b + 4ull * 128 * 256, b8 + 512,
            hx, 256, 256, g4, S, nullptr, 0, 0, ACT(1, 1, 2, 1));
        k_lstm<<<gEW, blk, 0, stream>>>(g4, cst, hz);
        // conv1..3: [384,128] x Hmid(hz ch0..127, kstride 256) -> vh, kh, qh_raw
        k_gemm<<<dim3(128, 3), blk, 0, stream>>>(w5b, b5,
            hz, 128, 256, att, S, nullptr, 0, 0, ACT(0, 0, 0, 0));
        k_tq<<<2048, blk, 0, stream>>>(att + 2 * S, qt);
        // conv4..5: [256,128] x m -> km, vm
        k_gemm<<<dim3(128, 2), blk, 0, stream>>>(w5b + 3ull * 128 * 128, b5 + 384,
            mbf, 128, 128, att + 3 * S, S, nullptr, 0, 0, ACT(0, 0, 0, 0));
        k_soft<<<dim3(16, 32, 4), blk, 0, stream>>>(att, qt, zbf);
        // conv6: [128,256] x concat(zh,zm) -> z (bf16 into hz ch128..255)
        k_gemm<<<dim3(128, 1), blk, 0, stream>>>(w8b, b8,
            zbf, 256, 256, nullptr, 0, hz, 256, 128, ACT(0, 0, 0, 0));
        // conv7..9: [384,256] x hz -> ot(sig), gt(tanh), it(sig)  (reuse g4)
        k_gemm<<<dim3(128, 3), blk, 0, stream>>>(w8b + 1ull * 128 * 256, b8 + 128,
            hz, 256, 256, g4, S, nullptr, 0, 0, ACT(1, 2, 1, 0));
        k_final<<<gEW, blk, 0, stream>>>(g4, mfp, x, t + 1, hx, mbf,
            (t == 15) ? out : nullptr);
    }
}
